// NAO_53609781789397
// MI455X (gfx1250) — compile-verified, matrix operand order checked by reading, not run on hardware
//
#include <hip/hip_runtime.h>
#include <cstdint>

typedef __attribute__((ext_vector_type(16))) _Float16 v16h;
typedef __attribute__((ext_vector_type(8)))  _Float16 v8h;
typedef __attribute__((ext_vector_type(16))) __bf16   v16bf;
typedef __attribute__((ext_vector_type(8)))  __bf16   v8bf;
typedef __attribute__((ext_vector_type(8)))  float    v8f;
typedef __attribute__((ext_vector_type(4)))  float    v4f;
typedef __attribute__((ext_vector_type(4)))  unsigned int u32x4;
typedef __attribute__((ext_vector_type(8)))  int      i32x8;
typedef __attribute__((ext_vector_type(4)))  int      i32x4;

#define DEVINL static __device__ __forceinline__

constexpr int TT  = 1600;   // NTOKEN
constexpr int T2  = 3200;   // 2*NTOKEN
constexpr int FF  = 128;    // FEAT
constexpr int HH  = 8;      // HEAD
constexpr int HD  = 512;    // HEAD*DK
constexpr int NB  = 8;      // batch
constexpr int KC  = 160;    // K-chunk staged by TDM (divides 3200 and 1600)
constexpr float EPS_   = 1e-5f;
constexpr float INV39  = 1.0f / 39.0f;
constexpr float DXDY_  = INV39 * INV39;
constexpr float SCALE_ = DXDY_ * 0.125f;   // DXDY / sqrt(64)

DEVINL float lrelu(float x) { return fmaxf(x, 0.01f * x); }   // branch-free

DEVINL v8f wmma_bf(v16bf a, v16bf b, v8f c) {
  return __builtin_amdgcn_wmma_f32_16x16x32_bf16(false, a, false, b, (short)0, c, false, false);
}
DEVINL v8f wmma_h(v16h a, v16h b, v8f c) {
  return __builtin_amdgcn_wmma_f32_16x16x32_f16(false, a, false, b, (short)0, c, false, false);
}

// 16-bit WMMA operand for K-range [kBase, kBase+32): lane's elements are two
// contiguous 8-half chunks at +8*hf and +16+8*hf -> two 128-bit loads.
DEVINL v16bf load_op_bf(const __bf16* base, int hf) {
  const v8bf lo = *reinterpret_cast<const v8bf*>(base + 8 * hf);
  const v8bf hi = *reinterpret_cast<const v8bf*>(base + 16 + 8 * hf);
  return __builtin_shufflevector(lo, hi, 0, 1, 2, 3, 4, 5, 6, 7,
                                 8, 9, 10, 11, 12, 13, 14, 15);
}
DEVINL v16h load_op_h(const _Float16* base, int hf) {
  const v8h lo = *reinterpret_cast<const v8h*>(base + 8 * hf);
  const v8h hi = *reinterpret_cast<const v8h*>(base + 16 + 8 * hf);
  return __builtin_shufflevector(lo, hi, 0, 1, 2, 3, 4, 5, 6, 7,
                                 8, 9, 10, 11, 12, 13, 14, 15);
}

// ---------------------------------------------------------------------------
// TDM: async 2-D tile load (bf16) global -> LDS.  D# per ISA 08_async_tensor §8.
// tile = tile_rows x tile_k elements, row stride = row_stride elements,
// packed contiguously into LDS at lds_off.  (6-arg builtin on this toolchain.)
// ---------------------------------------------------------------------------
DEVINL void tdm_load_2d(unsigned lds_off, const void* gaddr, unsigned tile_k,
                        unsigned tile_rows, unsigned row_stride) {
  const unsigned long long ga = (unsigned long long)(uintptr_t)gaddr;
  u32x4 g0;
  g0[0] = 1u;                                        // count=1, user descriptor
  g0[1] = lds_off;                                   // lds_addr (bytes)
  g0[2] = (unsigned)(ga & 0xFFFFFFFFu);              // global_addr[31:0]
  g0[3] = (unsigned)((ga >> 32) & 0x01FFFFFFu) | (2u << 30);  // addr[56:32] | type=2
  i32x8 g1;
  g1[0] = (1 << 16);                                 // data_size = 2 bytes
  g1[1] = (int)((row_stride & 0xFFFFu) << 16);       // tensor_dim0[15:0]
  g1[2] = (int)(((row_stride >> 16) & 0xFFFFu) |     // tensor_dim0[31:16]
                ((tile_rows & 0xFFFFu) << 16));      // tensor_dim1[15:0]
  g1[3] = (int)((tile_k & 0xFFFFu) << 16);           // tile_dim0
  g1[4] = (int)(tile_rows & 0xFFFFu);                // tile_dim1 (tile_dim2=0)
  g1[5] = (int)row_stride;                           // tensor_dim0_stride[31:0]
  g1[6] = 0;                                         // stride hi / dim1_stride lo
  g1[7] = 0;
  const i32x4 z4 = {0, 0, 0, 0};
  const i32x8 z8 = {0, 0, 0, 0, 0, 0, 0, 0};
  __builtin_amdgcn_tensor_load_to_lds(g0, g1, z4, z4, z8, 0);
}

// ---------------------------------------------------------------------------
// Kernel 1: whole-(2T,F) LayerNorm per batch; emit V f32, V bf16, V^T bf16,
// xy[:T]^T bf16.
// ---------------------------------------------------------------------------
__global__ void ln0_kernel(const float* __restrict__ xy,
                           const float* __restrict__ g0, const float* __restrict__ b0,
                           float* __restrict__ Vf, __bf16* __restrict__ Vbf,
                           __bf16* __restrict__ Vt, __bf16* __restrict__ xyt) {
  const int b = blockIdx.x;
  const int N = T2 * FF;
  const float* x = xy + (size_t)b * N;
  __shared__ float s1[1024], s2[1024];
  float sum = 0.f, ss = 0.f;
  for (int i = threadIdx.x; i < N; i += blockDim.x) { float v = x[i]; sum += v; ss += v * v; }
  s1[threadIdx.x] = sum; s2[threadIdx.x] = ss; __syncthreads();
  for (int st = 512; st > 0; st >>= 1) {
    if ((int)threadIdx.x < st) { s1[threadIdx.x] += s1[threadIdx.x + st]; s2[threadIdx.x] += s2[threadIdx.x + st]; }
    __syncthreads();
  }
  const float mu = s1[0] / (float)N;
  const float var = s2[0] / (float)N - mu * mu;
  const float rs = rsqrtf(var + EPS_);
  for (int i = threadIdx.x; i < N; i += blockDim.x) {
    const int t = i / FF, f = i % FF;
    float v = (x[i] - mu) * rs * g0[i] + b0[i];
    Vf[(size_t)b * N + i] = v;
    const __bf16 vb = (__bf16)v;
    Vbf[(size_t)b * N + i] = vb;
    Vt[(size_t)b * FF * T2 + (size_t)f * T2 + t] = vb;
  }
  for (int i = threadIdx.x; i < TT * FF; i += blockDim.x) {
    const int t = i / FF, f = i % FF;
    xyt[(size_t)b * FF * TT + (size_t)f * TT + t] = (__bf16)x[i];
  }
}

// ---------------------------------------------------------------------------
// Kernel 2a/2b: weight packing (transposed, low precision)
// ---------------------------------------------------------------------------
__global__ void pack_wqk_kernel(const float* __restrict__ Wq, const float* __restrict__ Wk,
                                __bf16* __restrict__ WqT, __bf16* __restrict__ WkT) {
  const int tid = blockIdx.x * blockDim.x + threadIdx.x;   // 4*512*128
  const int j = tid / (HD * FF);
  const int r = tid % (HD * FF);
  const int n = r / FF, f = r % FF;
  const size_t src = ((size_t)(j * HH + (n >> 6)) * FF + f) * 64 + (n & 63);
  WqT[tid] = (__bf16)Wq[src];
  WkT[tid] = (__bf16)Wk[src];
}
__global__ void pack_mlp_kernel(const float* __restrict__ kW2, const float* __restrict__ fW2,
                                _Float16* __restrict__ kW2T, _Float16* __restrict__ fW2T) {
  const int tid = blockIdx.x * blockDim.x + threadIdx.x;   // 2048
  const int c = tid / 32, k = tid % 32;
  kW2T[c * 32 + k] = (_Float16)kW2[k * 64 + c];
  fW2T[c * 32 + k] = (_Float16)fW2[k * 64 + c];
}

// ---------------------------------------------------------------------------
// Kernel 3: edge MLPs -> W2 (T x 2T) bf16.  One wave = 16 edges x both MLPs.
// ---------------------------------------------------------------------------
__global__ void edge_kernel(const float* __restrict__ kW1, const float* __restrict__ kb1,
                            const _Float16* __restrict__ kW2T, const float* __restrict__ kb2,
                            const float* __restrict__ kW3, const float* __restrict__ kb3,
                            const float* __restrict__ fW1, const float* __restrict__ fb1,
                            const _Float16* __restrict__ fW2T, const float* __restrict__ fb2,
                            const float* __restrict__ fW3, const float* __restrict__ fb3,
                            __bf16* __restrict__ W2out) {
  const int wid = blockIdx.x * (blockDim.x >> 5) + (threadIdx.x >> 5);
  const int ln = threadIdx.x & 31;
  const int hf = ln >> 4, mr = ln & 15;
  const int base = wid * 16;
  const int E = base + mr;
  const int i = E / TT, j = E % TT;
  const float f0 = (float)(i / 40) * INV39;
  const float f1 = (float)(i % 40) * INV39;
  const float f2 = (float)(j / 40) * INV39;
  const float f3 = (float)(j % 40) * INV39;
  const int row = base / TT;
  const int j0  = base % TT;

  const float*    W1s[2] = {kW1, fW1};   const float* b1s[2] = {kb1, fb1};
  const _Float16* W2s[2] = {kW2T, fW2T}; const float* b2s[2] = {kb2, fb2};
  const float*    W3s[2] = {kW3, fW3};   const float* b3s[2] = {kb3, fb3};

  for (int p = 0; p < 2; ++p) {
    const float* W1 = W1s[p];     const float* b1 = b1s[p];
    const _Float16* W2T = W2s[p]; const float* b2 = b2s[p];
    const float* W3 = W3s[p];     const float* b3 = b3s[p];
    v16h a;
#pragma unroll
    for (int g = 0; g < 4; ++g) {            // elems 4g..4g+3 in operand K-order
      const int koff = (g < 2) ? (8 * hf + 4 * g) : (16 + 8 * hf + 4 * (g - 2));
      v4f h4 = *reinterpret_cast<const v4f*>(b1 + koff);
      h4 += f0 * *reinterpret_cast<const v4f*>(W1 + koff);
      h4 += f1 * *reinterpret_cast<const v4f*>(W1 + 32 + koff);
      h4 += f2 * *reinterpret_cast<const v4f*>(W1 + 64 + koff);
      h4 += f3 * *reinterpret_cast<const v4f*>(W1 + 96 + koff);
#pragma unroll
      for (int t = 0; t < 4; ++t) a[4 * g + t] = (_Float16)lrelu(h4[t]);
    }
    float part[8];
#pragma unroll
    for (int v = 0; v < 8; ++v) part[v] = 0.f;
#pragma unroll
    for (int n = 0; n < 4; ++n) {
      const int col = n * 16 + mr;
      v16h bb = load_op_h(W2T + col * 32, hf);
      v8f acc = {};
      acc = wmma_h(a, bb, acc);
      const float bias = b2[col];
      const float w3 = W3[col];
#pragma unroll
      for (int v = 0; v < 8; ++v) part[v] += lrelu(acc[v] + bias) * w3;
    }
#pragma unroll
    for (int off = 1; off < 16; off <<= 1)
#pragma unroll
      for (int v = 0; v < 8; ++v) part[v] += __shfl_xor(part[v], off, 32);
    if (mr == 0) {
      const float b3v = b3[0];
      v8bf o;
#pragma unroll
      for (int v = 0; v < 8; ++v) o[v] = (__bf16)(part[v] + b3v);
      *reinterpret_cast<v8bf*>(W2out + (size_t)row * T2 + p * TT + j0 + 8 * hf) = o;
    }
  }
}

// ---------------------------------------------------------------------------
// Kernel 4: Q/K projection, 32x64 tile/wave, K=128.
// store_t=0: Q row-major; store_t=1: K^T (vector stores).
// ---------------------------------------------------------------------------
__global__ void qk_kernel(const __bf16* __restrict__ Vbf, const __bf16* __restrict__ WT,
                          const float* __restrict__ bias, __bf16* __restrict__ Out,
                          int store_t) {
  const int wid = blockIdx.x * (blockDim.x >> 5) + (threadIdx.x >> 5);
  const int ln = threadIdx.x & 31, hf = ln >> 4, mr = ln & 15;
  const int b = wid / (100 * 8);
  const int r = wid % (100 * 8);
  const int rowBase = (r / 8) * 32, colBase = (r % 8) * 64;
  const __bf16* A = Vbf + (size_t)b * T2 * FF;
  v8f acc[2][4] = {};
  for (int kk = 0; kk < FF; kk += 32) {
    v16bf a0 = load_op_bf(A + (size_t)(rowBase + mr) * FF + kk, hf);
    v16bf a1 = load_op_bf(A + (size_t)(rowBase + 16 + mr) * FF + kk, hf);
#pragma unroll
    for (int n = 0; n < 4; ++n) {
      const int col = colBase + n * 16 + mr;
      v16bf bb = load_op_bf(WT + (size_t)col * FF + kk, hf);
      acc[0][n] = wmma_bf(a0, bb, acc[0][n]);
      acc[1][n] = wmma_bf(a1, bb, acc[1][n]);
    }
  }
  if (store_t) {
    __bf16* O = Out + (size_t)b * HD * T2;   // [n][t]
#pragma unroll
    for (int s = 0; s < 2; ++s)
#pragma unroll
      for (int n = 0; n < 4; ++n) {
        const int col = colBase + n * 16 + mr;
        const float bv = bias[col];
        v8bf o;
#pragma unroll
        for (int v = 0; v < 8; ++v) o[v] = (__bf16)(acc[s][n][v] + bv);
        *reinterpret_cast<v8bf*>(O + (size_t)col * T2 + rowBase + 16 * s + 8 * hf) = o;
      }
  } else {
    __bf16* O = Out + (size_t)b * T2 * HD;   // [t][n]
#pragma unroll
    for (int s = 0; s < 2; ++s)
#pragma unroll
      for (int n = 0; n < 4; ++n) {
        const int col = colBase + n * 16 + mr;
        const float bv = bias[col];
#pragma unroll
        for (int v = 0; v < 8; ++v)
          O[(size_t)(rowBase + 16 * s + v + 8 * hf) * HD + col] = (__bf16)(acc[s][n][v] + bv);
      }
  }
}

// ---------------------------------------------------------------------------
// Kernel 5: KV = scale * K^T @ Vside.  Block = 4 waves sharing a TDM-staged
// 64x160 B tile in LDS (double buffered).  Out: KV^T [f][n].
// ---------------------------------------------------------------------------
__global__ void __launch_bounds__(128) kv_kernel(const __bf16* __restrict__ Kt,
                          const __bf16* __restrict__ VT, __bf16* __restrict__ KVt,
                          int nchunk, size_t strideVT, int ldvt, float scale) {
  __shared__ __attribute__((aligned(16))) __bf16 sB[2][64 * KC];
  const int ln = threadIdx.x & 31, wv = threadIdx.x >> 5;
  const int hf = ln >> 4, mr = ln & 15;
  const int b = blockIdx.x / 16;
  const int r = blockIdx.x % 16;
  const int rowBase = (r / 2) * 64 + wv * 16;      // n (0..511)
  const int colBase = (r % 2) * 64;                // f
  const __bf16* Ab = Kt + (size_t)b * HD * T2 + (size_t)(rowBase + mr) * T2;
  const __bf16* Bb = VT + (size_t)b * strideVT + (size_t)colBase * ldvt;
  if (wv == 0)
    tdm_load_2d((unsigned)(uintptr_t)&sB[0][0], Bb, KC, 64, (unsigned)ldvt);
  v8f acc[4] = {};
  for (int c = 0; c < nchunk; ++c) {
    const int cur = c & 1;
    if (wv == 0) __builtin_amdgcn_s_wait_tensorcnt(0);
    __syncthreads();
    if (wv == 0 && c + 1 < nchunk)
      tdm_load_2d((unsigned)(uintptr_t)&sB[cur ^ 1][0], Bb + (size_t)(c + 1) * KC,
                  KC, 64, (unsigned)ldvt);
    const int kk0 = c * KC;
#pragma unroll
    for (int ks = 0; ks < KC / 32; ++ks) {
      v16bf a = load_op_bf(Ab + kk0 + ks * 32, hf);
#pragma unroll
      for (int n = 0; n < 4; ++n) {
        v16bf bb = load_op_bf(&sB[cur][(n * 16 + mr) * KC + ks * 32], hf);
        acc[n] = wmma_bf(a, bb, acc[n]);
      }
    }
    __syncthreads();
  }
  __bf16* O = KVt + (size_t)b * FF * HD;           // [f][n]
#pragma unroll
  for (int n = 0; n < 4; ++n) {
    const int col = colBase + n * 16 + mr;
    v8bf o;
#pragma unroll
    for (int v = 0; v < 8; ++v) o[v] = (__bf16)(acc[n][v] * scale);
    *reinterpret_cast<v8bf*>(O + (size_t)col * HD + rowBase + 8 * hf) = o;
  }
}

// ---------------------------------------------------------------------------
// Kernel 6a (layers 0..2): C[f][t] = KV^T x Q^T; the D-layout vector store
// emits mid[t][f] row-major f32 directly (2x b128 per tile).
// ---------------------------------------------------------------------------
__global__ void midA_kernel(const __bf16* __restrict__ KVt, const __bf16* __restrict__ Qbf,
                            float* __restrict__ midf) {
  const int wid = blockIdx.x * (blockDim.x >> 5) + (threadIdx.x >> 5);
  const int ln = threadIdx.x & 31, hf = ln >> 4, mr = ln & 15;
  const int b = wid / (4 * 50);
  const int r = wid % (4 * 50);
  const int rowBase = (r / 50) * 32;               // f
  const int colBase = (r % 50) * 64;               // t
  const __bf16* A = KVt + (size_t)b * FF * HD;
  const __bf16* Bb = Qbf + (size_t)b * T2 * HD;
  v8f acc[2][4] = {};
  for (int kk = 0; kk < HD; kk += 32) {
    v16bf a0 = load_op_bf(A + (size_t)(rowBase + mr) * HD + kk, hf);
    v16bf a1 = load_op_bf(A + (size_t)(rowBase + 16 + mr) * HD + kk, hf);
#pragma unroll
    for (int n = 0; n < 4; ++n) {
      const int col = colBase + n * 16 + mr;       // t
      v16bf bb = load_op_bf(Bb + (size_t)col * HD + kk, hf);
      acc[0][n] = wmma_bf(a0, bb, acc[0][n]);
      acc[1][n] = wmma_bf(a1, bb, acc[1][n]);
    }
  }
#pragma unroll
  for (int s = 0; s < 2; ++s)
#pragma unroll
    for (int n = 0; n < 4; ++n) {
      const int t = colBase + n * 16 + mr;
      *reinterpret_cast<v8f*>(midf + ((size_t)b * T2 + t) * FF + rowBase + 16 * s + 8 * hf)
          = acc[s][n];
    }
}

// ---------------------------------------------------------------------------
// Kernel 6b (final layer): mid^T [f][t] bf16 = (Q @ KV)^T via vector stores.
// ---------------------------------------------------------------------------
__global__ void midT_kernel(const __bf16* __restrict__ Qbf, const __bf16* __restrict__ KVt,
                            __bf16* __restrict__ midT) {
  const int wid = blockIdx.x * (blockDim.x >> 5) + (threadIdx.x >> 5);
  const int ln = threadIdx.x & 31, hf = ln >> 4, mr = ln & 15;
  const int b = wid / (100 * 2);
  const int r = wid % (100 * 2);
  const int rowBase = (r / 2) * 32, colBase = (r % 2) * 64;
  const __bf16* A = Qbf + (size_t)b * T2 * HD;
  const __bf16* Bb = KVt + (size_t)b * FF * HD;
  v8f acc[2][4] = {};
  for (int kk = 0; kk < HD; kk += 32) {
    v16bf a0 = load_op_bf(A + (size_t)(rowBase + mr) * HD + kk, hf);
    v16bf a1 = load_op_bf(A + (size_t)(rowBase + 16 + mr) * HD + kk, hf);
#pragma unroll
    for (int n = 0; n < 4; ++n) {
      const int col = colBase + n * 16 + mr;
      v16bf bb = load_op_bf(Bb + (size_t)col * HD + kk, hf);
      acc[0][n] = wmma_bf(a0, bb, acc[0][n]);
      acc[1][n] = wmma_bf(a1, bb, acc[1][n]);
    }
  }
  __bf16* O = midT + (size_t)b * FF * T2;          // [f][t]
#pragma unroll
  for (int s = 0; s < 2; ++s)
#pragma unroll
    for (int n = 0; n < 4; ++n) {
      const int col = colBase + n * 16 + mr;
      v8bf o;
#pragma unroll
      for (int v = 0; v < 8; ++v) o[v] = (__bf16)acc[s][n][v];
      *reinterpret_cast<v8bf*>(O + (size_t)col * T2 + rowBase + 16 * s + 8 * hf) = o;
    }
}

// ---------------------------------------------------------------------------
// Kernel 7: per-token LayerNorm + residual; refresh V f32, bf16, V^T.
// ---------------------------------------------------------------------------
__global__ void lnres_kernel(const float* __restrict__ midf, float* __restrict__ Vf,
                             __bf16* __restrict__ Vbf, __bf16* __restrict__ Vt,
                             const float* __restrict__ gl, const float* __restrict__ bl) {
  const int wid = blockIdx.x * (blockDim.x >> 5) + (threadIdx.x >> 5);
  const int ln = threadIdx.x & 31;
  const int b = wid / T2, t = wid % T2;
  const size_t rowoff = (size_t)wid * FF;
  const float* mrow = midf + rowoff;
  float x[4], sum = 0.f, ss = 0.f;
#pragma unroll
  for (int q = 0; q < 4; ++q) { x[q] = mrow[ln + 32 * q]; sum += x[q]; ss += x[q] * x[q]; }
#pragma unroll
  for (int off = 1; off < 32; off <<= 1) { sum += __shfl_xor(sum, off, 32); ss += __shfl_xor(ss, off, 32); }
  const float mu = sum / (float)FF;
  const float var = ss / (float)FF - mu * mu;
  const float rs = rsqrtf(var + EPS_);
  float* vrow = Vf + rowoff;
  __bf16* vbrow = Vbf + rowoff;
  __bf16* vtb = Vt + (size_t)b * FF * T2;
#pragma unroll
  for (int q = 0; q < 4; ++q) {
    const int f = ln + 32 * q;
    const float y = (x[q] - mu) * rs * gl[f] + bl[f] + vrow[f];
    vrow[f] = y;
    const __bf16 yb = (__bf16)y;
    vbrow[f] = yb;
    vtb[(size_t)f * T2 + t] = yb;
  }
}

// ---------------------------------------------------------------------------
// Kernel 8: out = DXDY * W2 @ mid.  Block = 4 waves (64 rows) sharing a
// TDM-staged 64x160 mid^T tile (double buffered).  K = 3200.
// ---------------------------------------------------------------------------
__global__ void __launch_bounds__(128) final_kernel(const __bf16* __restrict__ W2bf,
                             const __bf16* __restrict__ midT, float* __restrict__ out) {
  __shared__ __attribute__((aligned(16))) __bf16 sB[2][64 * KC];
  const int ln = threadIdx.x & 31, wv = threadIdx.x >> 5;
  const int hf = ln >> 4, mr = ln & 15;
  const int b = blockIdx.x / 50;
  const int r = blockIdx.x % 50;
  const int rowBase = (r / 2) * 64 + wv * 16;      // token row (0..1599)
  const int colBase = (r % 2) * 64;                // f
  const __bf16* Ab = W2bf + (size_t)(rowBase + mr) * T2;
  const __bf16* Bb = midT + (size_t)b * FF * T2 + (size_t)colBase * T2;
  if (wv == 0)
    tdm_load_2d((unsigned)(uintptr_t)&sB[0][0], Bb, KC, 64, (unsigned)T2);
  v8f acc[4] = {};
  for (int c = 0; c < T2 / KC; ++c) {
    const int cur = c & 1;
    if (wv == 0) __builtin_amdgcn_s_wait_tensorcnt(0);
    __syncthreads();
    if (wv == 0 && c + 1 < T2 / KC)
      tdm_load_2d((unsigned)(uintptr_t)&sB[cur ^ 1][0], Bb + (size_t)(c + 1) * KC,
                  KC, 64, (unsigned)T2);
    const int kk0 = c * KC;
#pragma unroll
    for (int ks = 0; ks < KC / 32; ++ks) {
      v16bf a = load_op_bf(Ab + kk0 + ks * 32, hf);
#pragma unroll
      for (int n = 0; n < 4; ++n) {
        v16bf bb = load_op_bf(&sB[cur][(n * 16 + mr) * KC + ks * 32], hf);
        acc[n] = wmma_bf(a, bb, acc[n]);
      }
    }
    __syncthreads();
  }
#pragma unroll
  for (int n = 0; n < 4; ++n) {
    const int col = colBase + n * 16 + mr;
#pragma unroll
    for (int v = 0; v < 8; ++v)
      out[((size_t)b * TT + rowBase + v + 8 * hf) * FF + col] = DXDY_ * acc[n][v];
  }
}

// ---------------------------------------------------------------------------
extern "C" void kernel_launch(void* const* d_in, const int* in_sizes, int n_in,
                              void* d_out, int out_size, void* d_ws, size_t ws_size,
                              hipStream_t stream) {
  (void)in_sizes; (void)n_in; (void)out_size; (void)ws_size;
  const float* xy  = (const float*)d_in[0];
  const float* kW1 = (const float*)d_in[1];  const float* kb1 = (const float*)d_in[2];
  const float* kW2 = (const float*)d_in[3];  const float* kb2 = (const float*)d_in[4];
  const float* kW3 = (const float*)d_in[5];  const float* kb3 = (const float*)d_in[6];
  const float* fW1 = (const float*)d_in[7];  const float* fb1 = (const float*)d_in[8];
  const float* fW2 = (const float*)d_in[9];  const float* fb2 = (const float*)d_in[10];
  const float* fW3 = (const float*)d_in[11]; const float* fb3 = (const float*)d_in[12];
  const float* Wq  = (const float*)d_in[13]; const float* bq  = (const float*)d_in[14];
  const float* Wk  = (const float*)d_in[15]; const float* bk  = (const float*)d_in[16];
  const float* g0  = (const float*)d_in[17]; const float* b0  = (const float*)d_in[18];
  const float* gl  = (const float*)d_in[19]; const float* bl  = (const float*)d_in[20];

  char* ws = (char*)d_ws;
  size_t off = 0;
  auto carve = [&](size_t bytes) -> char* {
    char* p = ws + off;
    off += (bytes + 255) & ~(size_t)255;
    return p;
  };
  __bf16*  W2bf = (__bf16*) carve((size_t)TT * T2 * 2);
  float*   Vf   = (float*)  carve((size_t)NB * T2 * FF * 4);
  __bf16*  Vbf  = (__bf16*) carve((size_t)NB * T2 * FF * 2);
  __bf16*  Vt   = (__bf16*) carve((size_t)NB * FF * T2 * 2);
  __bf16*  Qbf  = (__bf16*) carve((size_t)NB * T2 * HD * 2);
  __bf16*  Kt   = (__bf16*) carve((size_t)NB * HD * T2 * 2);
  __bf16*  KVt  = (__bf16*) carve((size_t)NB * FF * HD * 2);
  float*   midf = (float*)  carve((size_t)NB * T2 * FF * 4);
  __bf16*  midT = (__bf16*) carve((size_t)NB * FF * T2 * 2);
  __bf16*  xyt  = (__bf16*) carve((size_t)NB * FF * TT * 2);
  __bf16*  WqT  = (__bf16*) carve((size_t)4 * HD * FF * 2);
  __bf16*  WkT  = (__bf16*) carve((size_t)4 * HD * FF * 2);
  _Float16* kW2T = (_Float16*)carve((size_t)64 * 32 * 2);
  _Float16* fW2T = (_Float16*)carve((size_t)64 * 32 * 2);

  ln0_kernel<<<NB, 1024, 0, stream>>>(xy, g0, b0, Vf, Vbf, Vt, xyt);
  pack_wqk_kernel<<<1024, 256, 0, stream>>>(Wq, Wk, WqT, WkT);
  pack_mlp_kernel<<<8, 256, 0, stream>>>(kW2, fW2, kW2T, fW2T);
  edge_kernel<<<20000, 256, 0, stream>>>(kW1, kb1, kW2T, kb2, kW3, kb3,
                                         fW1, fb1, fW2T, fb2, fW3, fb3, W2bf);

  for (int j = 0; j < 4; ++j) {
    const __bf16* WqTj = WqT + (size_t)j * HD * FF;
    const __bf16* WkTj = WkT + (size_t)j * HD * FF;
    const float*  bqj  = bq + (size_t)j * HH * 64;
    const float*  bkj  = bk + (size_t)j * HH * 64;
    qk_kernel<<<1600, 128, 0, stream>>>(Vbf, WqTj, bqj, Qbf, 0);  // Q row-major
    qk_kernel<<<1600, 128, 0, stream>>>(Vbf, WkTj, bkj, Kt, 1);   // K^T
    if (j < 3) {
      kv_kernel<<<128, 128, 0, stream>>>(Kt, Vt, KVt, T2 / KC, (size_t)FF * T2, T2, SCALE_);
      midA_kernel<<<400, 128, 0, stream>>>(KVt, Qbf, midf);
      lnres_kernel<<<3200, 256, 0, stream>>>(midf, Vf, Vbf, Vt, gl + j * FF, bl + j * FF);
    } else {
      kv_kernel<<<128, 128, 0, stream>>>(Kt, xyt, KVt, TT / KC, (size_t)FF * TT, TT, SCALE_);
      midT_kernel<<<400, 128, 0, stream>>>(Qbf, KVt, midT);
      final_kernel<<<400, 128, 0, stream>>>(W2bf, midT, (float*)d_out);
    }
  }
}
